// MultiHeadAttentionLayer_12360915877953
// MI455X (gfx1250) — matrix operand order repaired and checked by reading, not compiled, not run on hardware
//
#include <hip/hip_runtime.h>
#include <hip/hip_bf16.h>
#include <type_traits>

typedef __bf16 bf16_t;
typedef __attribute__((ext_vector_type(16))) __bf16 v16bf;
typedef __attribute__((ext_vector_type(8)))  __bf16 v8bf;
typedef __attribute__((ext_vector_type(8)))  float  v8f;

#define EMBED    1024
#define HEADS    16
#define HEAD_DIM 64
#define BATCH    4
#define SEQ      2048
#define BS       (BATCH*SEQ)

union AFrag { v16bf v; v8bf h[2]; };

static __device__ __forceinline__ v8f wmma_bf16(const v16bf& a, const v16bf& b, const v8f& c) {
    return __builtin_amdgcn_wmma_f32_16x16x32_bf16(false, a, false, b, (short)0, c, false, false);
}

// ---------------------------------------------------------------- fp32 -> bf16
__global__ __launch_bounds__(256) void cvt_f32_bf16(const float* __restrict__ src,
                                                    bf16_t* __restrict__ dst, int n) {
    int i = blockIdx.x * blockDim.x + threadIdx.x;
    if (i < n) dst[i] = (bf16_t)src[i];
}

// ---------------------------------------------------------------- GEMM: out = X @ W^T + b
// X: [BS, EMBED] (fp32 or bf16), W: [EMBED, EMBED] bf16 row-major (W[n][k])
// OUT_MODE 0: bf16 [B,H,S,Dh]   (q, k)
// OUT_MODE 1: bf16 [B,H,Dh,S]   (v transposed, for contiguous attention loads)
// OUT_MODE 2: fp32 [B,S,E]      (final output)
template<bool IN_BF16, int OUT_MODE>
__global__ __launch_bounds__(256)
void gemm_wmma(const float* __restrict__ Xf, const bf16_t* __restrict__ Xb,
               const bf16_t* __restrict__ W, const float* __restrict__ bias,
               float* __restrict__ outF, bf16_t* __restrict__ outB)
{
    const int lane = threadIdx.x & 31;
    const int wid  = threadIdx.x >> 5;
    const int hi   = lane >> 4;       // half-wave select
    const int ln   = lane & 15;
    const int wm   = wid & 3;         // 4 waves in M
    const int wn   = wid >> 2;        // 2 waves in N
    const int Mbase = blockIdx.x * 64  + wm * 16;
    const int Nbase = blockIdx.y * 128 + wn * 64;

    v8f acc[4] = {};
    const int arow = Mbase + ln;      // A-layout: lane = row

    for (int kk = 0; kk < EMBED; kk += 32) {
        // A fragment 16x32: elem e -> K = hi*16 + e
        AFrag a;
        if constexpr (IN_BF16) {
            const bf16_t* p = Xb + (size_t)arow * EMBED + kk + hi * 16;
            a.h[0] = *(const v8bf*)(p);
            a.h[1] = *(const v8bf*)(p + 8);
            __builtin_prefetch(p + 32, 0, 1);               // next k-slab
        } else {
            const float* p = Xf + (size_t)arow * EMBED + kk + hi * 16;
#pragma unroll
            for (int e = 0; e < 8; ++e) {
                a.h[0][e] = (bf16_t)p[e];
                a.h[1][e] = (bf16_t)p[e + 8];
            }
            __builtin_prefetch(p + 32, 0, 1);               // next k-slab
        }
        // B fragment 32x16: elem e -> K = hi*16 + e, N = ln ; B[k][n] = W[n][k]
#pragma unroll
        for (int t = 0; t < 4; ++t) {
            const bf16_t* wp = W + (size_t)(Nbase + t * 16 + ln) * EMBED + kk + hi * 16;
            v16bf b = *(const v16bf*)wp;
            acc[t] = wmma_bf16(a.v, b, acc[t]);
        }
    }

    // C layout: VGPR r, lanes 0-15 -> M=r, lanes 16-31 -> M=r+8; N = ln
#pragma unroll
    for (int t = 0; t < 4; ++t) {
        const int n  = Nbase + t * 16 + ln;
        const float bv = bias[n];
#pragma unroll
        for (int r = 0; r < 8; ++r) {
            const int M   = Mbase + r + 8 * hi;
            const float v = acc[t][r] + bv;
            if constexpr (OUT_MODE == 2) {
                outF[(size_t)M * EMBED + n] = v;
            } else {
                const int b_ = M / SEQ, s = M % SEQ;
                const int h_ = n >> 6,  d = n & 63;
                size_t addr;
                if constexpr (OUT_MODE == 0)
                    addr = (((size_t)b_ * HEADS + h_) * SEQ + s) * HEAD_DIM + d;
                else
                    addr = (((size_t)b_ * HEADS + h_) * HEAD_DIM + d) * SEQ + s;
                outB[addr] = (bf16_t)v;
            }
        }
    }
}

// ---------------------------------------------------------------- causal flash attention
// q,k: [B,H,S,Dh] bf16 ; vT: [B,H,Dh,S] bf16 ; ctx: [B,S,E] bf16
// One wave = one (b,h, 16-query tile); key tiles of 32.
// TRANSPOSED formulation: S^T = K.Q^T  (C layout: lane = query, VGPRs = keys)
//   -> softmax stats per query live per-lane: in-lane trees + one shfl_xor(16).
//   O^T = V^T.P^T accumulated with dims as rows, queries as lanes
//   -> alpha / 1/l scaling is elementwise, no LDS staging, packed 16B ctx stores.
__global__ __launch_bounds__(128)
void flash_attn(const bf16_t* __restrict__ q, const bf16_t* __restrict__ k,
                const bf16_t* __restrict__ vT, bf16_t* __restrict__ ctx)
{
    const int lane = threadIdx.x & 31;
    const int wid  = threadIdx.x >> 5;
    const int hi   = lane >> 4;
    const int ln   = lane & 15;                 // = query column for this lane
    const int gw   = blockIdx.x * 4 + wid;      // global wave-tile id
    const int bh   = gw >> 7;                   // / (SEQ/16)
    const int qt   = gw & 127;
    const int qbase = qt * 16;

    const bf16_t* qb = q  + (size_t)bh * SEQ * HEAD_DIM;
    const bf16_t* kb = k  + (size_t)bh * SEQ * HEAD_DIM;
    const bf16_t* vb = vT + (size_t)bh * HEAD_DIM * SEQ;

    // Q as B-fragments (B[k=dim][n=query]): lane ln = query, elems = dims hi*16+e
    v16bf qfrag[2];
#pragma unroll
    for (int t = 0; t < 2; ++t)
        qfrag[t] = *(const v16bf*)(qb + (size_t)(qbase + ln) * HEAD_DIM + t * 32 + hi * 16);

    v8f o[4] = {};                              // O^T: rows = dims t*16 + r+8*hi, cols = query ln
    float mq = -1e30f, lq = 0.0f;               // per-query (per-lane) softmax state
    const float scale = 0.125f;                 // 1/sqrt(64); folded into exp args

    auto key_tile = [&](auto maskC, int jj) {
        constexpr bool MASK = decltype(maskC)::value;
        // ---- S^T tiles: s0 = keys jj..jj+15, s1 = keys jj+16..jj+31 (rows), queries (cols)
        v8f s0 = {}, s1 = {};
#pragma unroll
        for (int t = 0; t < 2; ++t) {
            AFrag ka;  // A[m=key][k=dim]: lane = key row
            const bf16_t* p0 = kb + (size_t)(jj + ln) * HEAD_DIM + t * 32 + hi * 16;
            ka.h[0] = *(const v8bf*)(p0);
            ka.h[1] = *(const v8bf*)(p0 + 8);
            s0 = wmma_bf16(ka.v, qfrag[t], s0);
            const bf16_t* p1 = kb + (size_t)(jj + 16 + ln) * HEAD_DIM + t * 32 + hi * 16;
            ka.h[0] = *(const v8bf*)(p1);
            ka.h[1] = *(const v8bf*)(p1 + 8);
            s1 = wmma_bf16(ka.v, qfrag[t], s1);
        }
        if constexpr (!MASK) {
            __builtin_prefetch(kb + (size_t)(jj + 32 + ln) * HEAD_DIM, 0, 1);
            __builtin_prefetch(kb + (size_t)(jj + 48 + ln) * HEAD_DIM, 0, 1);
#pragma unroll
            for (int t = 0; t < 4; ++t)
                __builtin_prefetch(vb + (size_t)(t * 16 + ln) * SEQ + jj + 32, 0, 1);
        }

        // lane (ln,hi) holds, for query ln: s0 -> keys jj + r+8*hi ; s1 -> keys jj+16 + r+8*hi
        float a0[8], a1[8];
#pragma unroll
        for (int r = 0; r < 8; ++r) { a0[r] = s0[r]; a1[r] = s1[r]; }
        if constexpr (MASK) {
            const int d = qbase + ln - jj - 8 * hi;   // key r masked iff r > d (resp r+16 > d)
#pragma unroll
            for (int r = 0; r < 8; ++r) {
                if (r > d)      a0[r] = -1e30f;
                if (r + 16 > d) a1[r] = -1e30f;
            }
        }
        // ---- in-lane max over 16, then combine halves with one shfl_xor(16)
        float mx[8];
#pragma unroll
        for (int r = 0; r < 8; ++r) mx[r] = fmaxf(a0[r], a1[r]);
        float tmax = fmaxf(fmaxf(fmaxf(mx[0], mx[1]), fmaxf(mx[2], mx[3])),
                           fmaxf(fmaxf(mx[4], mx[5]), fmaxf(mx[6], mx[7])));
        tmax = fmaxf(tmax, __shfl_xor(tmax, 16, 32));
        const float mnew  = fmaxf(mq, tmax);
        const float alpha = __expf((mq - mnew) * scale);

        float p0[8], p1[8];
#pragma unroll
        for (int r = 0; r < 8; ++r) {
            p0[r] = __expf((a0[r] - mnew) * scale);
            p1[r] = __expf((a1[r] - mnew) * scale);
        }
        float sm[8];
#pragma unroll
        for (int r = 0; r < 8; ++r) sm[r] = p0[r] + p1[r];
        float tsum = ((sm[0] + sm[1]) + (sm[2] + sm[3])) + ((sm[4] + sm[5]) + (sm[6] + sm[7]));
        tsum += __shfl_xor(tsum, 16, 32);
        lq = lq * alpha + tsum;
        mq = mnew;
#pragma unroll
        for (int t = 0; t < 4; ++t)
#pragma unroll
            for (int r = 0; r < 8; ++r) o[t][r] *= alpha;   // alpha is per-query = per-lane

        // ---- build P^T B-fragment (B[k=key][n=query]; lane needs keys hi*16+0..15)
        // own: hi=0 has keys 0-7 (p0) & 16-23 (p1); hi=1 has keys 8-15 (p0) & 24-31 (p1)
        // swap the "wrong half" values across lane^16:
        v16bf pb;
#pragma unroll
        for (int r = 0; r < 8; ++r) {
            const float send = hi ? p0[r] : p1[r];
            const float recv = __shfl_xor(send, 16, 32);
            pb[r]     = (bf16_t)(hi ? recv  : p0[r]);       // elems 0-7  : keys hi*16 + 0-7
            pb[r + 8] = (bf16_t)(hi ? p1[r] : recv);        // elems 8-15 : keys hi*16 + 8-15
        }
        // ---- O^T += V^T . P^T   (A = V^T slab: lane = dim row, contiguous keys)
#pragma unroll
        for (int t = 0; t < 4; ++t) {
            AFrag va;
            const bf16_t* vp = vb + (size_t)(t * 16 + ln) * SEQ + jj + hi * 16;
            va.h[0] = *(const v8bf*)(vp);
            va.h[1] = *(const v8bf*)(vp + 8);
            o[t] = wmma_bf16(va.v, pb, o[t]);
        }
    };

    const int jj_mask = qbase & ~31;            // the single diagonal-straddling tile
    for (int jj = 0; jj < jj_mask; jj += 32)
        key_tile(std::false_type{}, jj);        // mask-free hot loop
    key_tile(std::true_type{}, jj_mask);        // masked tail

    // ---- epilogue: normalize (per-lane) and write ctx[b, s=qbase+ln, h*64+dim]
    // O^T tile t: rows = dims t*16 + r + 8*hi -> 8 contiguous dims = one packed 16B store
    const int b_ = bh >> 4, h_ = bh & 15;
    const float inv = 1.0f / lq;
    bf16_t* cp = ctx + ((size_t)b_ * SEQ + qbase + ln) * EMBED + h_ * HEAD_DIM + 8 * hi;
#pragma unroll
    for (int t = 0; t < 4; ++t) {
        v8bf pk;
#pragma unroll
        for (int r = 0; r < 8; ++r) pk[r] = (bf16_t)(o[t][r] * inv);
        *(v8bf*)(cp + t * 16) = pk;
    }
}

// ----------------------------------------------------------------
extern "C" void kernel_launch(void* const* d_in, const int* in_sizes, int n_in,
                              void* d_out, int out_size, void* d_ws, size_t ws_size,
                              hipStream_t stream) {
    const float* Q  = (const float*)d_in[0];
    const float* K  = (const float*)d_in[1];
    const float* V  = (const float*)d_in[2];
    const float* Wq = (const float*)d_in[3];
    const float* Wk = (const float*)d_in[4];
    const float* Wv = (const float*)d_in[5];
    const float* Wo = (const float*)d_in[6];
    const float* bq = (const float*)d_in[7];
    const float* bk = (const float*)d_in[8];
    const float* bv = (const float*)d_in[9];
    const float* bo = (const float*)d_in[10];
    float* out = (float*)d_out;

    char* ws = (char*)d_ws;
    size_t off = 0;
    bf16_t* Wqb = (bf16_t*)(ws + off); off += (size_t)EMBED * EMBED * 2;
    bf16_t* Wkb = (bf16_t*)(ws + off); off += (size_t)EMBED * EMBED * 2;
    bf16_t* Wvb = (bf16_t*)(ws + off); off += (size_t)EMBED * EMBED * 2;
    bf16_t* Wob = (bf16_t*)(ws + off); off += (size_t)EMBED * EMBED * 2;
    bf16_t* qp  = (bf16_t*)(ws + off); off += (size_t)BS * EMBED * 2;   // [B,H,S,Dh]
    bf16_t* kp  = (bf16_t*)(ws + off); off += (size_t)BS * EMBED * 2;   // [B,H,S,Dh]
    bf16_t* vp  = (bf16_t*)(ws + off); off += (size_t)BS * EMBED * 2;   // [B,H,Dh,S]
    bf16_t* ctx = (bf16_t*)(ws + off); off += (size_t)BS * EMBED * 2;   // [B,S,E]

    const int nw = EMBED * EMBED;
    dim3 cb(256), cg((nw + 255) / 256);
    cvt_f32_bf16<<<cg, cb, 0, stream>>>(Wq, Wqb, nw);
    cvt_f32_bf16<<<cg, cb, 0, stream>>>(Wk, Wkb, nw);
    cvt_f32_bf16<<<cg, cb, 0, stream>>>(Wv, Wvb, nw);
    cvt_f32_bf16<<<cg, cb, 0, stream>>>(Wo, Wob, nw);

    dim3 gb(256);
    dim3 gg(BS / 64, EMBED / 128);
    gemm_wmma<false, 0><<<gg, gb, 0, stream>>>(Q, nullptr, Wqb, bq, nullptr, qp);
    gemm_wmma<false, 0><<<gg, gb, 0, stream>>>(K, nullptr, Wkb, bk, nullptr, kp);
    gemm_wmma<false, 1><<<gg, gb, 0, stream>>>(V, nullptr, Wvb, bv, nullptr, vp);

    flash_attn<<<dim3(BATCH * HEADS * (SEQ / 16) / 4), dim3(128), 0, stream>>>(qp, kp, vp, ctx);

    gemm_wmma<true, 2><<<gg, gb, 0, stream>>>(nullptr, ctx, Wob, bo, out, nullptr);
}
